// TransformerEncoder_78185584657152
// MI455X (gfx1250) — matrix operand order repaired and checked by reading, not compiled, not run on hardware
//
#include <hip/hip_runtime.h>
#include <hip/hip_bf16.h>

// ---- CDNA5 WMMA types ----
typedef __attribute__((ext_vector_type(16))) __bf16        v16bf;
typedef __attribute__((ext_vector_type(8)))  float         v8f;
typedef __attribute__((ext_vector_type(4)))  unsigned int  u32x4;

#define NLAYER 12
#define BATCH  64
#define SEQ    197
#define CDIM   768
#define NHEAD  12
#define HDIM   64
#define FFDIM  3072
#define MROWS  (BATCH * SEQ)   // 12608 = 788*16 = 197*64
#define NPAD   224             // padded seq (multiple of 32)
#define SOFT_SCALE 0.125f      // 64^-0.5

union AFrag { v16bf v; u32x4 q[2]; };

__device__ inline v8f zero8() {
  v8f z = {0.f,0.f,0.f,0.f,0.f,0.f,0.f,0.f};
  return z;
}

// A-fragment (16x32, M x K): lane holds row m=lane&15; two 16B chunks at
// k + (lane>>4)*8 and k + 16 + (lane>>4)*8  (ISA 7.12.2, 16-bit A layout)
__device__ inline v16bf load_a(const __bf16* base, int ld, int row, int k, int lane) {
  const __bf16* p = base + (size_t)row * ld + k + ((lane >> 4) << 3);
  AFrag f;
  f.q[0] = *(const u32x4*)(p);
  f.q[1] = *(const u32x4*)(p + 16);
  return f.v;
}

// B-fragment (32x16, K x N): lane holds col n=lane&15; two 16B chunks at
// k + (lane>>4)*8 and k + 16 + (lane>>4)*8 (same K order as A)
__device__ inline v16bf load_b(const __bf16* base, int ld, int row, int k, int lane) {
  const __bf16* p = base + (size_t)row * ld + k + ((lane >> 4) << 3);
  AFrag f;
  f.q[0] = *(const u32x4*)(p);
  f.q[1] = *(const u32x4*)(p + 16);
  return f.v;
}

__device__ inline v8f wmma_bf16(v16bf a, v16bf b, v8f c) {
  // (neg_a, A, neg_b, B, c_mod, C, reuse_a, reuse_b)
  return __builtin_amdgcn_wmma_f32_16x16x32_bf16(false, a, false, b, (short)0, c,
                                                 false, false);
}

// --------------------------------------------------------------------------
// Generic WMMA GEMM: out[M, Nout] = A[M,K](bf16) @ W[Nout,K]^T(bf16) + epilogue
// block: 256 thr = 8 waves; block tile 64(M) x 128(N); wave tile 16 x 64.
// EPI 0: qkv scatter (+q/v bias) -> padded q/k/v^T bf16 buffers
// EPI 1: hbuf[m,n] += gamma[n] * (acc + bias0[n])
// EPI 2: obf[m,n]  = bf16(gelu(acc + bias0[n]))
// --------------------------------------------------------------------------
template <int EPI>
__global__ void __launch_bounds__(256)
gemm_kernel(const __bf16* __restrict__ A, const __bf16* __restrict__ W,
            int K, int Nout,
            const float* __restrict__ bias0, const float* __restrict__ bias1,
            const float* __restrict__ gamma, float* __restrict__ hbuf,
            __bf16* __restrict__ obf,
            __bf16* __restrict__ qb, __bf16* __restrict__ kb, __bf16* __restrict__ vb)
{
  const int tid  = threadIdx.x;
  const int lane = tid & 31;
  const int wave = tid >> 5;
  const int wm   = wave & 3;        // 4 waves stacked in M
  const int wn   = wave >> 2;       // 2 wave columns in N
  const int bm   = blockIdx.x * 64;
  const int bn   = blockIdx.y * 128;
  const int col0 = lane & 15;
  const int half = lane >> 4;

  const int arow = bm + wm * 16 + col0;
  const int bnb  = bn + wn * 64;

  v8f acc[4];
#pragma unroll
  for (int t = 0; t < 4; ++t) acc[t] = zero8();

  for (int k = 0; k < K; k += 32) {
    v16bf a = load_a(A, K, arow, k, lane);
#pragma unroll
    for (int t = 0; t < 4; ++t) {
      v16bf b = load_b(W, K, bnb + t * 16 + col0, k, lane);
      acc[t] = wmma_bf16(a, b, acc[t]);
    }
  }

#pragma unroll
  for (int t = 0; t < 4; ++t) {
    const int nbase = bnb + t * 16;           // multiple of 16
    if (EPI == 0) {
      // kind/head/d-base are uniform across the 16-wide tile (768, 64 | 16)
      const int kind = nbase / CDIM;          // 0=q 1=k 2=v  (uniform)
      const int c0   = nbase - kind * CDIM;
      const int hh   = c0 >> 6;               // uniform head
      const int d    = (c0 & 63) + col0;
      const int c    = c0 + col0;
      const float bv = (kind == 0) ? bias0[c] : ((kind == 2) ? bias1[c] : 0.f);
#pragma unroll
      for (int r = 0; r < 8; ++r) {
        const int m  = bm + wm * 16 + r + half * 8;
        const int bb = m / SEQ;
        const int nn = m - bb * SEQ;
        const __bf16 o = (__bf16)(acc[t][r] + bv);
        if (kind == 0)
          qb[(((size_t)(bb * NHEAD + hh)) * NPAD + nn) * HDIM + d] = o;
        else if (kind == 1)
          kb[(((size_t)(bb * NHEAD + hh)) * NPAD + nn) * HDIM + d] = o;
        else  // V stored transposed: [B,H,64,224] so P@V B-frags are contiguous
          vb[(((size_t)(bb * NHEAD + hh)) * HDIM + d) * NPAD + nn] = o;
      }
    } else if (EPI == 1) {
      const int n = nbase + col0;
      const float gb = gamma[n];
      const float b0 = bias0[n];
#pragma unroll
      for (int r = 0; r < 8; ++r) {
        const int m = bm + wm * 16 + r + half * 8;
        const size_t idx = (size_t)m * CDIM + n;
        hbuf[idx] += gb * (acc[t][r] + b0);
      }
    } else {
      const int n = nbase + col0;
      const float b0 = bias0[n];
#pragma unroll
      for (int r = 0; r < 8; ++r) {
        const int m = bm + wm * 16 + r + half * 8;
        const float x = acc[t][r] + b0;
        const float g = 0.5f * x * (1.f + erff(x * 0.70710678118654752f));
        obf[(size_t)m * Nout + n] = (__bf16)g;
      }
    }
  }
}

// --------------------------------------------------------------------------
// Fused attention per (b, h, 32-row q tile):
//   S = scale*q k^T + padded relbias (WMMA -> LDS f32; mask baked into bias),
//   8-thread/row softmax, P (bf16, padded to 224) in LDS,
//   O = P @ V (WMMA, A-fragments from LDS via ds_load)
// --------------------------------------------------------------------------
__global__ void __launch_bounds__(256)
attn_kernel(const __bf16* __restrict__ qbuf, const __bf16* __restrict__ kbuf,
            const __bf16* __restrict__ vbuf, const float* __restrict__ biaspad,
            __bf16* __restrict__ obf)
{
  __shared__ __align__(16) float  S[32][208];
  __shared__ __align__(16) __bf16 P[32][NPAD];

  const int tid  = threadIdx.x;
  const int lane = tid & 31;
  const int wave = tid >> 5;
  const int col0 = lane & 15;
  const int half = lane >> 4;
  const int qt = blockIdx.x, h = blockIdx.y, b = blockIdx.z;
  const int qm0 = qt * 32;

  const __bf16* qp = qbuf + (size_t)(b * NHEAD + h) * NPAD * HDIM;
  const __bf16* kp = kbuf + (size_t)(b * NHEAD + h) * NPAD * HDIM;
  const __bf16* vp = vbuf + (size_t)(b * NHEAD + h) * HDIM * NPAD;
  const float*  bp = biaspad + (size_t)h * NPAD * NPAD;   // [224][224], mask baked in

  // ---- stage 1: scores. 26 jobs = 13 n-tiles x 2 m-subtiles ----
  for (int j = wave; j < 26; j += 8) {
    const int sm = j & 1;
    const int nt = j >> 1;
    v8f acc = zero8();
    const int arow = qm0 + sm * 16 + col0;
    const int brow = nt * 16 + col0;
#pragma unroll
    for (int kk = 0; kk < HDIM; kk += 32) {
      v16bf a  = load_a(qp, HDIM, arow, kk, lane);
      v16bf bb = load_b(kp, HDIM, brow, kk, lane);
      acc = wmma_bf16(a, bb, acc);
    }
    const int nl = nt * 16 + col0;           // 0..207
#pragma unroll
    for (int r = 0; r < 8; ++r) {
      const int ml = sm * 16 + r + half * 8; // 0..31
      const int qn = qm0 + ml;               // 0..223 (padded bias covers all)
      S[ml][nl] = acc[r] * SOFT_SCALE + bp[(size_t)qn * NPAD + nl];
    }
  }
  __syncthreads();

  // ---- stage 2: row softmax, 8 threads per row (shfl_xor within wave) ----
  {
    const int row = tid >> 3;     // 0..31 (rows map to consecutive 8-lane groups)
    const int sub = tid & 7;
    float mx = -1e30f;
    for (int c = sub; c < 208; c += 8) mx = fmaxf(mx, S[row][c]);
#pragma unroll
    for (int msk = 1; msk < 8; msk <<= 1) mx = fmaxf(mx, __shfl_xor(mx, msk, 32));
    float sum = 0.f;
    for (int c = sub; c < 208; c += 8) {
      const float e = __expf(S[row][c] - mx);
      S[row][c] = e;
      sum += e;
    }
#pragma unroll
    for (int msk = 1; msk < 8; msk <<= 1) sum += __shfl_xor(sum, msk, 32);
    const float inv = 1.f / sum;
    for (int c = sub; c < NPAD; c += 8)
      P[row][c] = (c < 208) ? (__bf16)(S[row][c] * inv) : (__bf16)0.f;
  }
  __syncthreads();

  // ---- stage 3: O = P @ V. 8 jobs = 2 m-subtiles x 4 d-tiles, 1/wave ----
  {
    const int sm = wave & 1;
    const int dt = wave >> 1;
    v8f acc = zero8();
    const int arow = sm * 16 + col0;           // LDS row
    const int brow = dt * 16 + col0;           // d index (v stored transposed)
#pragma unroll
    for (int kk = 0; kk < NPAD; kk += 32) {
      v16bf a  = load_a(&P[0][0], NPAD, arow, kk, lane);   // ds_load path
      v16bf bb = load_b(vp, NPAD, brow, kk, lane);
      acc = wmma_bf16(a, bb, acc);
    }
#pragma unroll
    for (int r = 0; r < 8; ++r) {
      const int ml = sm * 16 + r + half * 8;
      const int qn = qm0 + ml;
      if (qn < SEQ) {
        const int d = dt * 16 + col0;
        obf[(size_t)(b * SEQ + qn) * CDIM + h * HDIM + d] = (__bf16)acc[r];
      }
    }
  }
}

// --------------------------------------------------------------------------
// LayerNorm + cast to bf16: one block per row (C=768 = 256*3)
// --------------------------------------------------------------------------
__global__ void __launch_bounds__(256)
ln_kernel(const float* __restrict__ hbuf, const float* __restrict__ sc,
          const float* __restrict__ bi, __bf16* __restrict__ zbf)
{
  __shared__ float red[256];
  const int row = blockIdx.x;
  const int tid = threadIdx.x;
  const float* x = hbuf + (size_t)row * CDIM;
  const float v0 = x[tid], v1 = x[tid + 256], v2 = x[tid + 512];

  red[tid] = v0 + v1 + v2;
  __syncthreads();
  for (int off = 128; off > 0; off >>= 1) {
    if (tid < off) red[tid] += red[tid + off];
    __syncthreads();
  }
  const float mean = red[0] * (1.f / CDIM);
  __syncthreads();

  const float d0 = v0 - mean, d1 = v1 - mean, d2 = v2 - mean;
  red[tid] = d0 * d0 + d1 * d1 + d2 * d2;
  __syncthreads();
  for (int off = 128; off > 0; off >>= 1) {
    if (tid < off) red[tid] += red[tid + off];
    __syncthreads();
  }
  const float rs = rsqrtf(red[0] * (1.f / CDIM) + 1e-5f);

  __bf16* z = zbf + (size_t)row * CDIM;
  z[tid]       = (__bf16)(d0 * rs * sc[tid]       + bi[tid]);
  z[tid + 256] = (__bf16)(d1 * rs * sc[tid + 256] + bi[tid + 256]);
  z[tid + 512] = (__bf16)(d2 * rs * sc[tid + 512] + bi[tid + 512]);
}

// f32 -> bf16 weight staging
__global__ void cvt_kernel(const float* __restrict__ src, __bf16* __restrict__ dst, int n) {
  int i = blockIdx.x * blockDim.x + threadIdx.x;
  const int stride = gridDim.x * blockDim.x;
  for (; i < n; i += stride) dst[i] = (__bf16)src[i];
}

// Padded bias: biaspad[h][qn][kn], qn/kn in [0,224):
//   kn>=197 -> -1e30 (key mask); qn>=197 -> 0 (unused row); else table lookup.
__global__ void relbias_kernel(const float* __restrict__ tab, const int* __restrict__ idx,
                               float* __restrict__ out) {
  const int NN = NPAD * NPAD;
  const int i = blockIdx.x * blockDim.x + threadIdx.x;
  if (i < NHEAD * NN) {
    const int h   = i / NN;
    const int rem = i - h * NN;
    const int qn  = rem / NPAD;
    const int kn  = rem - qn * NPAD;
    float v;
    if (kn >= SEQ)      v = -1e30f;
    else if (qn >= SEQ) v = 0.f;
    else                v = tab[idx[qn * SEQ + kn] * NHEAD + h];
    out[i] = v;
  }
}

extern "C" void kernel_launch(void* const* d_in, const int* in_sizes, int n_in,
                              void* d_out, int out_size, void* d_ws, size_t ws_size,
                              hipStream_t stream)
{
  (void)in_sizes; (void)n_in; (void)out_size; (void)ws_size;

  const float* x        = (const float*)d_in[0];
  const float* qkv_w    = (const float*)d_in[1];
  const float* q_bias   = (const float*)d_in[2];
  const float* v_bias   = (const float*)d_in[3];
  const float* proj_w   = (const float*)d_in[4];
  const float* proj_b   = (const float*)d_in[5];
  const float* ln1_s    = (const float*)d_in[6];
  const float* ln1_b    = (const float*)d_in[7];
  const float* ln2_s    = (const float*)d_in[8];
  const float* ln2_b    = (const float*)d_in[9];
  const float* fc1_w    = (const float*)d_in[10];
  const float* fc1_b    = (const float*)d_in[11];
  const float* fc2_w    = (const float*)d_in[12];
  const float* fc2_b    = (const float*)d_in[13];
  const float* gamma1   = (const float*)d_in[14];
  const float* gamma2   = (const float*)d_in[15];
  const float* rel_tab  = (const float*)d_in[16];
  const int*   rel_idx  = (const int*)d_in[17];

  // ---- workspace carve-up (all offsets 256B aligned) ----
  char* ws = (char*)d_ws;
  size_t off = 0;
  auto alloc = [&](size_t bytes) -> void* {
    void* p = (void*)(ws + off);
    off += (bytes + 255) & ~(size_t)255;
    return p;
  };
  const size_t qkv_elems = (size_t)BATCH * NHEAD * NPAD * HDIM;  // 11,010,048
  float*  hbuf  = (float*) alloc((size_t)MROWS * CDIM * 4);
  __bf16* zbf   = (__bf16*)alloc((size_t)MROWS * CDIM * 2);
  __bf16* qb    = (__bf16*)alloc(qkv_elems * 2);
  __bf16* kb    = (__bf16*)alloc(qkv_elems * 2);
  __bf16* vb    = (__bf16*)alloc(qkv_elems * 2);
  __bf16* mlp   = (__bf16*)alloc((size_t)MROWS * FFDIM * 2);
  __bf16* wqkv  = (__bf16*)alloc((size_t)3 * CDIM * CDIM * 2);
  __bf16* wproj = (__bf16*)alloc((size_t)CDIM * CDIM * 2);
  __bf16* wfc1  = (__bf16*)alloc((size_t)FFDIM * CDIM * 2);
  __bf16* wfc2  = (__bf16*)alloc((size_t)CDIM * FFDIM * 2);
  float*  relb  = (float*) alloc((size_t)NHEAD * NPAD * NPAD * 4);

  // residual stream starts as x
  hipMemcpyAsync(hbuf, x, (size_t)MROWS * CDIM * 4, hipMemcpyDeviceToDevice, stream);
  // zero padded q/k/v once per launch (deterministic, graph-legal)
  hipMemsetAsync(qb, 0, qkv_elems * 2, stream);
  hipMemsetAsync(kb, 0, qkv_elems * 2, stream);
  hipMemsetAsync(vb, 0, qkv_elems * 2, stream);

  relbias_kernel<<<(NHEAD * NPAD * NPAD + 255) / 256, 256, 0, stream>>>(rel_tab, rel_idx, relb);

  for (int l = 0; l < NLAYER; ++l) {
    const float* qw  = qkv_w  + (size_t)l * 3 * CDIM * CDIM;
    const float* pw  = proj_w + (size_t)l * CDIM * CDIM;
    const float* f1w = fc1_w  + (size_t)l * FFDIM * CDIM;
    const float* f2w = fc2_w  + (size_t)l * CDIM * FFDIM;

    cvt_kernel<<<4096, 256, 0, stream>>>(qw,  wqkv,  3 * CDIM * CDIM);
    cvt_kernel<<<2048, 256, 0, stream>>>(pw,  wproj, CDIM * CDIM);
    cvt_kernel<<<4096, 256, 0, stream>>>(f1w, wfc1,  FFDIM * CDIM);
    cvt_kernel<<<4096, 256, 0, stream>>>(f2w, wfc2,  CDIM * FFDIM);

    // LN1 -> zbf (bf16)
    ln_kernel<<<MROWS, 256, 0, stream>>>(hbuf, ln1_s + l * CDIM, ln1_b + l * CDIM, zbf);

    // QKV GEMM: [12608,2304] — scatter to padded q/k/v^T
    gemm_kernel<0><<<dim3(MROWS / 64, (3 * CDIM) / 128), 256, 0, stream>>>(
        zbf, wqkv, CDIM, 3 * CDIM,
        q_bias + l * CDIM, v_bias + l * CDIM,
        nullptr, nullptr, nullptr, qb, kb, vb);

    // fused attention -> zbf (bf16, [B,N,C])
    attn_kernel<<<dim3(7, NHEAD, BATCH), 256, 0, stream>>>(qb, kb, vb, relb, zbf);

    // proj GEMM + gamma1 residual into hbuf
    gemm_kernel<1><<<dim3(MROWS / 64, CDIM / 128), 256, 0, stream>>>(
        zbf, wproj, CDIM, CDIM,
        proj_b + l * CDIM, nullptr, gamma1 + l * CDIM, hbuf,
        nullptr, nullptr, nullptr, nullptr);

    // LN2 -> zbf
    ln_kernel<<<MROWS, 256, 0, stream>>>(hbuf, ln2_s + l * CDIM, ln2_b + l * CDIM, zbf);

    // FC1 GEMM + GELU -> mlp (bf16)
    gemm_kernel<2><<<dim3(MROWS / 64, FFDIM / 128), 256, 0, stream>>>(
        zbf, wfc1, CDIM, FFDIM,
        fc1_b + l * FFDIM, nullptr, nullptr, nullptr,
        mlp, nullptr, nullptr, nullptr);

    // FC2 GEMM + gamma2 residual into hbuf
    gemm_kernel<1><<<dim3(MROWS / 64, CDIM / 128), 256, 0, stream>>>(
        mlp, wfc2, FFDIM, CDIM,
        fc2_b + l * CDIM, nullptr, gamma2 + l * CDIM, hbuf,
        nullptr, nullptr, nullptr, nullptr);
  }

  hipMemcpyAsync(d_out, hbuf, (size_t)MROWS * CDIM * 4, hipMemcpyDeviceToDevice, stream);
}
